// ImprovedChannelAttentionModule_90065464197446
// MI455X (gfx1250) — matrix operand order repaired and checked by reading, not compiled, not run on hardware
//
#include <hip/hip_runtime.h>
#include <hip/hip_fp16.h>
#include <math.h>

#define B_   32
#define C_   512
#define H_   64
#define W_   64
#define HID_ 32
#define NREG 5
#define ROWS (NREG * B_)   // 160
#define BC_  (B_ * C_)     // 16384

typedef __attribute__((ext_vector_type(16))) _Float16 v16h;
typedef __attribute__((ext_vector_type(8)))  float    v8f;
typedef __attribute__((ext_vector_type(4)))  float    v4f;

// ---------------------------------------------------------------------------
// K1: per-(b,c) pooled stats. Quadrant avg/max + global avg/max.
// avg_all/max_all layout: [5, B, C] with region order
// [global, (0,0), (0,1), (1,0), (1,1)]  (quadrant q=(i<<1)|j -> slot q+1).
// x loads use default RT policy so lines stay resident in the 192MB L2
// for the second pass (apply_kernel).
// ---------------------------------------------------------------------------
__global__ void pool_kernel(const float* __restrict__ x,
                            float* __restrict__ avg_all,
                            float* __restrict__ max_all) {
  const int bc = blockIdx.x;                       // b*C + c
  const float* xp = x + (size_t)bc * (H_ * W_);
  const int t = threadIdx.x;                       // 0..255
  const int q = t >> 6;                            // quadrant 0..3
  const int l = t & 63;                            // lane within quadrant group
  const int qi = q >> 1, qj = q & 1;

  float s = 0.f, mx = -3.402823466e38f;
#pragma unroll
  for (int k = 0; k < 4; ++k) {
    const int f = l + 64 * k;                      // float4 slot 0..255 in quadrant
    const int row = qi * 32 + (f >> 3);
    const int col = qj * 32 + (f & 7) * 4;
    const float4 v = *(const float4*)(xp + row * W_ + col);
    s += v.x + v.y + v.z + v.w;
    mx = fmaxf(mx, fmaxf(fmaxf(v.x, v.y), fmaxf(v.z, v.w)));
  }

  __shared__ float ssum[256];
  __shared__ float smax[256];
  ssum[t] = s;
  smax[t] = mx;
  __syncthreads();
  for (int off = 32; off > 0; off >>= 1) {
    if (l < off) {
      ssum[t] += ssum[t + off];
      smax[t] = fmaxf(smax[t], smax[t + off]);
    }
    __syncthreads();
  }
  if (t == 0) {
    float gs = 0.f, gm = -3.402823466e38f;
#pragma unroll
    for (int qq = 0; qq < 4; ++qq) {
      const float qa = ssum[qq * 64] * (1.f / 1024.f);
      const float qM = smax[qq * 64];
      avg_all[(qq + 1) * BC_ + bc] = qa;
      max_all[(qq + 1) * BC_ + bc] = qM;
      gs += qa;
      gm = fmaxf(gm, qM);
    }
    avg_all[bc] = gs * 0.25f;
    max_all[bc] = gm;
  }
}

// ---------------------------------------------------------------------------
// K2a: Y = relu(A @ W1^T + b1).  A: [160,512], W1: [32,512], Y: [160,32].
// One wave per 16x16 output tile; blockIdx = (mtile, ntile, path).
// Fragment layouts per CDNA5 ISA 7.12.2 (16-bit A 16x32, B 32x16, f32 C/D).
// ---------------------------------------------------------------------------
__global__ void fc1_kernel(const float* __restrict__ avg_all,
                           const float* __restrict__ max_all,
                           const float* __restrict__ Wa1,
                           const float* __restrict__ ba1,
                           const float* __restrict__ Wm1,
                           const float* __restrict__ bm1,
                           float* __restrict__ Y_avg,
                           float* __restrict__ Y_max) {
  const int mt = blockIdx.x, nt = blockIdx.y, path = blockIdx.z;
  const float* A  = path ? max_all : avg_all;
  const float* W1 = path ? Wm1 : Wa1;
  const float* b1 = path ? bm1 : ba1;
  float* Y        = path ? Y_max : Y_avg;

  const int lane = threadIdx.x & 31;
  const int mn   = lane & 15;     // A row (within tile) / B col (within tile)
  const int half = lane >> 4;
  const int col  = nt * 16 + mn;  // output column n (0..31)
  const int rowA = mt * 16 + mn;  // A row m

  v8f acc;
  const float bias = b1[col];
#pragma unroll
  for (int r = 0; r < 8; ++r) acc[r] = bias;

  for (int kb = 0; kb < C_; kb += 32) {
    v16h a, b;
#pragma unroll
    for (int e = 0; e < 16; ++e) {
      const int v  = e >> 1;
      const int ka = ((v >= 4) ? 16 : 0) + (v & 3) * 2 + (e & 1) + half * 8;
      a[e] = (_Float16)A[rowA * C_ + kb + ka];
      const int kbv = ((e >= 8) ? 16 : 0) + (e & 7) + half * 8;
      b[e] = (_Float16)W1[col * C_ + kb + kbv];   // B(k,n) = W1[n,k]
    }
    acc = __builtin_amdgcn_wmma_f32_16x16x32_f16(
        false, a, false, b, (short)0, acc, false, false);
  }

#pragma unroll
  for (int r = 0; r < 8; ++r) {
    const float y = acc[r] > 0.f ? acc[r] : 0.f;   // ReLU
    const int row = mt * 16 + half * 8 + r;
    Y[row * HID_ + col] = y;
  }
}

// ---------------------------------------------------------------------------
// K2b: feats = (Y_avg @ Wa2^T + ba2) + (Y_max @ Wm2^T + bm2).
// Y: [160,32], W2: [512,32], feats: [160,512] == [5,B,C].
// K = 32 -> exactly one WMMA per path; both paths share one accumulator.
// ---------------------------------------------------------------------------
__global__ void fc2_kernel(const float* __restrict__ Y_avg,
                           const float* __restrict__ Y_max,
                           const float* __restrict__ Wa2,
                           const float* __restrict__ ba2,
                           const float* __restrict__ Wm2,
                           const float* __restrict__ bm2,
                           float* __restrict__ feats) {
  const int mt = blockIdx.x, nt = blockIdx.y;
  const int lane = threadIdx.x & 31;
  const int mn   = lane & 15;
  const int half = lane >> 4;
  const int col  = nt * 16 + mn;   // 0..511
  const int rowA = mt * 16 + mn;

  v8f acc;
  const float bias = ba2[col] + bm2[col];
#pragma unroll
  for (int r = 0; r < 8; ++r) acc[r] = bias;

#pragma unroll
  for (int path = 0; path < 2; ++path) {
    const float* Y  = path ? Y_max : Y_avg;
    const float* W2 = path ? Wm2 : Wa2;
    v16h a, b;
#pragma unroll
    for (int e = 0; e < 16; ++e) {
      const int v  = e >> 1;
      const int ka = ((v >= 4) ? 16 : 0) + (v & 3) * 2 + (e & 1) + half * 8;
      a[e] = (_Float16)Y[rowA * HID_ + ka];
      const int kbv = ((e >= 8) ? 16 : 0) + (e & 7) + half * 8;
      b[e] = (_Float16)W2[col * HID_ + kbv];      // B(k,n) = W2[n,k]
    }
    acc = __builtin_amdgcn_wmma_f32_16x16x32_f16(
        false, a, false, b, (short)0, acc, false, false);
  }

#pragma unroll
  for (int r = 0; r < 8; ++r) {
    const int row = mt * 16 + half * 8 + r;
    feats[row * C_ + col] = acc[r];
  }
}

// ---------------------------------------------------------------------------
// K3: fusion (softmax over 5 fusion weights, sigmoid gating) + elementwise
// rescale of x. One block per (b,c); float4 streaming.
// x loads are last-use -> non-temporal; out stores are write-once ->
// non-temporal, so the output stream does not evict L2-resident x lines.
// ---------------------------------------------------------------------------
__global__ void apply_kernel(const float* __restrict__ x,
                             const float* __restrict__ feats,
                             const float* __restrict__ fusion_w,
                             float* __restrict__ out) {
  const int bc = blockIdx.x;
  const size_t base = (size_t)bc * (H_ * W_);

  // softmax over the 5 fusion weights (uniform, scalar-load friendly)
  float fw[NREG], f[NREG];
  float wmax = -3.402823466e38f;
#pragma unroll
  for (int s = 0; s < NREG; ++s) {
    fw[s] = fusion_w[s];
    wmax = fmaxf(wmax, fw[s]);
  }
  float wsum = 0.f;
#pragma unroll
  for (int s = 0; s < NREG; ++s) {
    fw[s] = expf(fw[s] - wmax);
    wsum += fw[s];
  }
  const float winv = 1.f / wsum;

  float fused = 0.f;
#pragma unroll
  for (int s = 0; s < NREG; ++s) {
    f[s] = feats[s * BC_ + bc];
    fused += f[s] * fw[s] * winv;
  }
  const float ga = 1.f / (1.f + expf(-fused));

  const int t = threadIdx.x;
  const int q = t >> 6;
  const int l = t & 63;
  const int qi = q >> 1, qj = q & 1;
  const float scale = 1.f / (1.f + expf(-f[q + 1])) + ga;

#pragma unroll
  for (int k = 0; k < 4; ++k) {
    const int fidx = l + 64 * k;
    const int row = qi * 32 + (fidx >> 3);
    const int col = qj * 32 + (fidx & 7) * 4;
    const size_t idx = base + row * W_ + col;
    v4f v = __builtin_nontemporal_load((const v4f*)(x + idx));
    v.x *= scale; v.y *= scale; v.z *= scale; v.w *= scale;
    __builtin_nontemporal_store(v, (v4f*)(out + idx));
  }
}

// ---------------------------------------------------------------------------
extern "C" void kernel_launch(void* const* d_in, const int* in_sizes, int n_in,
                              void* d_out, int out_size, void* d_ws, size_t ws_size,
                              hipStream_t stream) {
  const float* x   = (const float*)d_in[0];
  const float* Wa1 = (const float*)d_in[1];
  const float* ba1 = (const float*)d_in[2];
  const float* Wa2 = (const float*)d_in[3];
  const float* ba2 = (const float*)d_in[4];
  const float* Wm1 = (const float*)d_in[5];
  const float* bm1 = (const float*)d_in[6];
  const float* Wm2 = (const float*)d_in[7];
  const float* bm2 = (const float*)d_in[8];
  const float* fw  = (const float*)d_in[9];
  float* out = (float*)d_out;

  float* ws      = (float*)d_ws;
  float* avg_all = ws;                        // [5,B,C] = 81920
  float* max_all = avg_all + ROWS * C_;       // [5,B,C] = 81920
  float* Y_avg   = max_all + ROWS * C_;       // [160,32] = 5120
  float* Y_max   = Y_avg + ROWS * HID_;       // [160,32] = 5120
  float* feats   = Y_max + ROWS * HID_;       // [160,512] = 81920
  // total: ~1.0 MB of d_ws

  pool_kernel<<<BC_, 256, 0, stream>>>(x, avg_all, max_all);
  fc1_kernel<<<dim3(ROWS / 16, HID_ / 16, 2), 32, 0, stream>>>(
      avg_all, max_all, Wa1, ba1, Wm1, bm1, Y_avg, Y_max);
  fc2_kernel<<<dim3(ROWS / 16, C_ / 16), 32, 0, stream>>>(
      Y_avg, Y_max, Wa2, ba2, Wm2, bm2, feats);
  apply_kernel<<<BC_, 256, 0, stream>>>(x, feats, fw, out);
}
